// MoE_predictor_61838939128387
// MI455X (gfx1250) — matrix operand order repaired and checked by reading, not compiled, not run on hardware
//
#include <hip/hip_runtime.h>
#include <hip/hip_bf16.h>
#include <math.h>

typedef __attribute__((ext_vector_type(16))) _Float16 v16h;
typedef __attribute__((ext_vector_type(8)))  _Float16 h8v;
typedef __attribute__((ext_vector_type(8)))  float    v8f;

#define TOK   3200      // B*S
#define H_    1024
#define E_    16
#define TWO_H 2048

__device__ __forceinline__ float gelu_f(float x) {
    return 0.5f * x * (1.0f + erff(x * 0.70710678118654752f));
}

__device__ __forceinline__ float block_sum(float v, float* red) {
    int tid = threadIdx.x;
    red[tid] = v;
    __syncthreads();
    for (int off = 128; off > 0; off >>= 1) {
        if (tid < off) red[tid] += red[tid + off];
        __syncthreads();
    }
    float r = red[0];
    __syncthreads();
    return r;
}

// ---------------------------------------------------------------------------
// Weight convert: w[e][k][n] f32 -> wt[e][n][k] f16 (transposed for WMMA B)
// ---------------------------------------------------------------------------
__global__ __launch_bounds__(256) void convT_kernel(const float* __restrict__ w,
                                                    _Float16* __restrict__ wt) {
    size_t i = (size_t)blockIdx.x * 256 + threadIdx.x;   // 16*1024*1024 total
    size_t e = i >> 20;
    size_t n = (i >> 10) & 1023;
    size_t k = i & 1023;
    wt[i] = (_Float16)w[(e << 20) + (k << 10) + n];
}

__global__ __launch_bounds__(256) void zero_kernel(float* __restrict__ p, int n) {
    int i = blockIdx.x * 256 + threadIdx.x;
    if (i < n) p[i] = 0.f;
}

// ---------------------------------------------------------------------------
// Embed: x = gelu(layernorm(cls_x)) -> f16 activations
// ---------------------------------------------------------------------------
__global__ __launch_bounds__(256) void embed_kernel(const float* __restrict__ xin,
                                                    const float* __restrict__ g,
                                                    const float* __restrict__ b,
                                                    _Float16* __restrict__ xh) {
    int t = blockIdx.x, tid = threadIdx.x;
    const float* xr = xin + (size_t)t * H_;
    __shared__ float red[256];
    float s1 = 0.f;
    for (int i = tid; i < H_; i += 256) s1 += xr[i];
    float mu = block_sum(s1, red) * (1.f / H_);
    float s2 = 0.f;
    for (int i = tid; i < H_; i += 256) { float d = xr[i] - mu; s2 += d * d; }
    float var = block_sum(s2, red) * (1.f / H_);
    float rstd = rsqrtf(var + 1e-5f);
    for (int i = tid; i < H_; i += 256)
        xh[(size_t)t * H_ + i] = (_Float16)gelu_f((xr[i] - mu) * rstd * g[i] + b[i]);
}

// ---------------------------------------------------------------------------
// Gate: probs = softmax(x @ gate_w + gate_b), top-2 -> scale[t][e]
// ---------------------------------------------------------------------------
__global__ __launch_bounds__(256) void gate_kernel(const _Float16* __restrict__ xh,
                                                   const float* __restrict__ gw,
                                                   const float* __restrict__ gb,
                                                   float* __restrict__ scale) {
    int t = blockIdx.x, tid = threadIdx.x;
    int e = tid & 15, gidx = tid >> 4;
    __shared__ float part[256];
    __shared__ float logit[16];
    const _Float16* xr = xh + (size_t)t * H_;
    float a = 0.f;
    for (int h = gidx; h < H_; h += 16) a += (float)xr[h] * gw[h * 16 + e];
    part[tid] = a;
    __syncthreads();
    if (tid < 16) {
        float s = gb[tid];
        for (int gg = 0; gg < 16; ++gg) s += part[gg * 16 + tid];
        logit[tid] = s;
    }
    __syncthreads();
    if (tid == 0) {
        float mx = logit[0];
        for (int i = 1; i < 16; ++i) mx = fmaxf(mx, logit[i]);
        float pr[16], ssum = 0.f;
        for (int i = 0; i < 16; ++i) { pr[i] = expf(logit[i] - mx); ssum += pr[i]; }
        float inv = 1.f / ssum;
        for (int i = 0; i < 16; ++i) pr[i] *= inv;
        int i1 = 0;
        for (int i = 1; i < 16; ++i) if (pr[i] > pr[i1]) i1 = i;
        int i2 = -1;
        for (int i = 0; i < 16; ++i) {
            if (i == i1) continue;
            if (i2 < 0 || pr[i] > pr[i2]) i2 = i;
        }
        for (int i = 0; i < 16; ++i)
            scale[(size_t)t * 16 + i] = (i == i1 || i == i2) ? pr[i] : 0.f;
    }
}

// ---------------------------------------------------------------------------
// Fused MoE expert block: 16 tokens x 1 expert, WMMA f16 -> f32.
//   GEMM1: H16 = gelu(X16 @ W1[e] + b1[e])       (16 x 1024, staged in LDS)
//   GEMM2: O16 = gelu(H16 @ W2[e] + b2[e])
//   out += gate_scale * O16   (atomic f32 accumulate)
// ---------------------------------------------------------------------------
__device__ __forceinline__ v16h make_frag(const _Float16* p0, const _Float16* p1) {
    h8v lo = *(const h8v*)p0;
    h8v hi = *(const h8v*)p1;
    return __builtin_shufflevector(lo, hi, 0, 1, 2, 3, 4, 5, 6, 7,
                                           8, 9, 10, 11, 12, 13, 14, 15);
}

__device__ __forceinline__ void wave_gemm16(const _Float16* __restrict__ As, // LDS 16x1024 row-major
                                            const _Float16* __restrict__ Bg, // global [n][k] f16
                                            int lane, int nbase, v8f* acc) {
    const int row  = lane & 15;
    const int hi16 = lane >> 4;          // 0 for lanes 0..15, 1 for lanes 16..31
    const int koffA = hi16 * 8;          // A: low lanes K 0..7/16..23, high lanes 8..15/24..31
    const int koffB = hi16 * 8;          // B: low lanes K 0..7/16..23, high lanes 8..15/24..31
    for (int k0 = 0; k0 < H_; k0 += 32) {
        const _Float16* pa = As + row * H_ + k0 + koffA;
        v16h afrag = make_frag(pa, pa + 16);
#pragma unroll
        for (int s = 0; s < 8; ++s) {
            const int n = nbase + s * 16 + row;
            const _Float16* pb = Bg + (size_t)n * H_ + k0 + koffB;
            __builtin_prefetch(pb + 32, 0, 1);  // global_prefetch_b8: next K chunk
            v16h bfrag = make_frag(pb, pb + 16);
            acc[s] = __builtin_amdgcn_wmma_f32_16x16x32_f16(
                false, afrag, false, bfrag, (short)0, acc[s], false, false);
        }
    }
}

__global__ __launch_bounds__(256) void moe_kernel(const _Float16* __restrict__ Xh,   // [TOK][1024]
                                                  const _Float16* __restrict__ W1T,  // [16][n][k]
                                                  const _Float16* __restrict__ W2T,  // [16][f][d]
                                                  const float* __restrict__ b1,     // [16][1024]
                                                  const float* __restrict__ b2,     // [16][1024]
                                                  const float* __restrict__ scale,  // [TOK][16]
                                                  float* __restrict__ out) {        // [TOK][1024] zeroed
    const int tile = blockIdx.x;   // 0..199
    const int e    = blockIdx.y;   // 0..15
    const int tid  = threadIdx.x, lane = tid & 31, wave = tid >> 5;
    extern __shared__ _Float16 smem[];
    _Float16* Xs = smem;               // 16 x 1024 f16
    _Float16* Hs = smem + 16 * H_;     // 16 x 1024 f16
    __shared__ float sc_row[16];

    if (tid < 16) sc_row[tid] = scale[(size_t)(tile * 16 + tid) * 16 + e];
    __syncthreads();
    bool any = false;
    for (int r = 0; r < 16; ++r) any |= (sc_row[r] != 0.f);
    if (!any) return;   // uniform exit: no token in this tile routed to expert e

    // stage 16x1024 activation tile
    const h8v* src = (const h8v*)(Xh + (size_t)tile * 16 * H_);
    for (int i = tid; i < 16 * H_ / 8; i += 256) ((h8v*)Xs)[i] = src[i];
    __syncthreads();

    const int nbase = wave * 128;
    const _Float16* B1 = W1T + ((size_t)e << 20);
    const _Float16* B2 = W2T + ((size_t)e << 20);
    const int hi16 = lane >> 4;
    const int col0 = lane & 15;

    // ---- GEMM1 ----
    v8f acc[8];
#pragma unroll
    for (int s = 0; s < 8; ++s) acc[s] = (v8f){0.f, 0.f, 0.f, 0.f, 0.f, 0.f, 0.f, 0.f};
    wave_gemm16(Xs, B1, lane, nbase, acc);
#pragma unroll
    for (int s = 0; s < 8; ++s) {
        const int col = nbase + s * 16 + col0;
        const float bb = b1[(size_t)e * H_ + col];
#pragma unroll
        for (int v = 0; v < 8; ++v) {
            const int rowM = v + hi16 * 8;            // C/D layout (ISA 7.12.2)
            Hs[rowM * H_ + col] = (_Float16)gelu_f(acc[s][v] + bb);
        }
    }
    __syncthreads();

    // ---- GEMM2 ----
#pragma unroll
    for (int s = 0; s < 8; ++s) acc[s] = (v8f){0.f, 0.f, 0.f, 0.f, 0.f, 0.f, 0.f, 0.f};
    wave_gemm16(Hs, B2, lane, nbase, acc);
#pragma unroll
    for (int s = 0; s < 8; ++s) {
        const int col = nbase + s * 16 + col0;
        const float bb = b2[(size_t)e * H_ + col];
#pragma unroll
        for (int v = 0; v < 8; ++v) {
            const int rowM = v + hi16 * 8;
            const float sc = sc_row[rowM];
            if (sc != 0.f) {
                const float val = gelu_f(acc[s][v] + bb);
                atomicAdd(&out[(size_t)(tile * 16 + rowM) * H_ + col], sc * val);
            }
        }
    }
}

// ---------------------------------------------------------------------------
// r = gelu(layernorm(moe_out))  (f32 -> f32 in d_out)
// ---------------------------------------------------------------------------
__global__ __launch_bounds__(256) void rout_kernel(const float* __restrict__ in,
                                                   const float* __restrict__ g,
                                                   const float* __restrict__ b,
                                                   float* __restrict__ out) {
    int t = blockIdx.x, tid = threadIdx.x;
    const float* xr = in + (size_t)t * H_;
    __shared__ float red[256];
    float s1 = 0.f;
    for (int i = tid; i < H_; i += 256) s1 += xr[i];
    float mu = block_sum(s1, red) * (1.f / H_);
    float s2 = 0.f;
    for (int i = tid; i < H_; i += 256) { float d = xr[i] - mu; s2 += d * d; }
    float var = block_sum(s2, red) * (1.f / H_);
    float rstd = rsqrtf(var + 1e-5f);
    for (int i = tid; i < H_; i += 256)
        out[(size_t)t * H_ + i] = gelu_f((xr[i] - mu) * rstd * g[i] + b[i]);
}

// ---------------------------------------------------------------------------
// c = gelu(layernorm(concat(r, r1)))  written transposed: ct[j][b][s]
// ---------------------------------------------------------------------------
__global__ __launch_bounds__(256) void concat_ln_kernel(const float* __restrict__ r0,
                                                        const float* __restrict__ r1,
                                                        const float* __restrict__ g,
                                                        const float* __restrict__ b,
                                                        float* __restrict__ ct) {
    int t = blockIdx.x, tid = threadIdx.x;
    int bb = t / 50, ss = t % 50;
    __shared__ float red[256];
    const float* a0 = r0 + (size_t)t * H_;
    const float* a1 = r1 + (size_t)t * H_;
    float s1 = 0.f;
    for (int i = tid; i < TWO_H; i += 256) {
        float v = (i < H_) ? a0[i] : a1[i - H_];
        s1 += v;
    }
    float mu = block_sum(s1, red) * (1.f / TWO_H);
    float s2 = 0.f;
    for (int i = tid; i < TWO_H; i += 256) {
        float v = (i < H_) ? a0[i] : a1[i - H_];
        float d = v - mu; s2 += d * d;
    }
    float var = block_sum(s2, red) * (1.f / TWO_H);
    float rstd = rsqrtf(var + 1e-5f);
    for (int i = tid; i < TWO_H; i += 256) {
        float v = (i < H_) ? a0[i] : a1[i - H_];
        ct[((size_t)i * 64 + bb) * 50 + ss] = gelu_f((v - mu) * rstd * g[i] + b[i]);
    }
}

// ---------------------------------------------------------------------------
// One MHA layer over c: [L=64][N=2048][E=50], 5 heads x 10, plus post MLP,
// residual. One block per n.
// ---------------------------------------------------------------------------
__global__ __launch_bounds__(256) void attn_kernel(const float* __restrict__ cin,
                                                   float* __restrict__ cout,
                                                   const float* __restrict__ wi,
                                                   const float* __restrict__ bi,
                                                   const float* __restrict__ wo,
                                                   const float* __restrict__ bo,
                                                   const float* __restrict__ pw,
                                                   const float* __restrict__ pb) {
    const int n = blockIdx.x, tid = threadIdx.x;
    extern __shared__ float sm[];
    float* xc = sm;            // 64*50
    float* qk = xc + 3200;     // 64*150
    float* sc = qk + 9600;     // 64*64
    float* ov = sc + 4096;     // 64*50
    const float* cb = cin + (size_t)n * 3200;
    for (int i = tid; i < 3200; i += 256) xc[i] = cb[i];
    __syncthreads();
    // qkv projection
    for (int i = tid; i < 64 * 150; i += 256) {
        int l = i / 150, f = i % 150;
        float a = bi[f];
        const float* wr = wi + f * 50;
        const float* xr = xc + l * 50;
        for (int e2 = 0; e2 < 50; ++e2) a += xr[e2] * wr[e2];
        qk[i] = a;
    }
    __syncthreads();
    const float inv = 0.316227766016838f;   // 1/sqrt(10)
    for (int h = 0; h < 5; ++h) {
        for (int i = tid; i < 4096; i += 256) {
            int l = i >> 6, m = i & 63;
            float a = 0.f;
            const float* qr = qk + l * 150 + h * 10;
            const float* kr = qk + m * 150 + 50 + h * 10;
            for (int d = 0; d < 10; ++d) a += qr[d] * kr[d];
            sc[i] = a * inv;
        }
        __syncthreads();
        if (tid < 64) {
            float* row = sc + tid * 64;
            float mx = -3.4e38f;
            for (int m = 0; m < 64; ++m) mx = fmaxf(mx, row[m]);
            float ssum = 0.f;
            for (int m = 0; m < 64; ++m) { float ee = expf(row[m] - mx); row[m] = ee; ssum += ee; }
            float rs = 1.f / ssum;
            for (int m = 0; m < 64; ++m) row[m] *= rs;
        }
        __syncthreads();
        for (int i = tid; i < 640; i += 256) {
            int l = i / 10, d = i % 10;
            float a = 0.f;
            const float* ar = sc + l * 64;
            for (int m = 0; m < 64; ++m) a += ar[m] * qk[m * 150 + 100 + h * 10 + d];
            ov[l * 50 + h * 10 + d] = a;
        }
        __syncthreads();
    }
    // output proj, gelu (reuse sc as [64][50] temp)
    for (int i = tid; i < 3200; i += 256) {
        int l = i / 50, f = i % 50;
        float a = bo[f];
        const float* wr = wo + f * 50;
        const float* orow = ov + l * 50;
        for (int e2 = 0; e2 < 50; ++e2) a += orow[e2] * wr[e2];
        sc[i] = gelu_f(a);
    }
    __syncthreads();
    float* co = cout + (size_t)n * 3200;
    for (int i = tid; i < 3200; i += 256) {
        int l = i / 50, f = i % 50;
        float a = pb[f];
        const float* wr = pw + f * 50;
        const float* grow = sc + l * 50;
        for (int e2 = 0; e2 < 50; ++e2) a += grow[e2] * wr[e2];
        co[i] = xc[i] + gelu_f(a);
    }
}

// ---------------------------------------------------------------------------
// Final: gather ct -> ln(aa)+gelu -> fusion matmul -> ln3+gelu -> mean over S
// ---------------------------------------------------------------------------
__global__ __launch_bounds__(256) void final_kernel(const float* __restrict__ ct,
                                                    const float* __restrict__ ag,
                                                    const float* __restrict__ ab,
                                                    const float* __restrict__ fw,
                                                    const float* __restrict__ fb,
                                                    const float* __restrict__ g3,
                                                    const float* __restrict__ b3,
                                                    float* __restrict__ result) {
    int t = blockIdx.x, tid = threadIdx.x;
    int bb = t / 50, ss = t % 50;
    extern __shared__ float sm[];
    float* cx = sm;            // 2048
    float* fx = sm + TWO_H;    // 1024
    __shared__ float red[256];
    float s1 = 0.f;
    for (int j = tid; j < TWO_H; j += 256) {
        float v = ct[((size_t)j * 64 + bb) * 50 + ss];
        cx[j] = v; s1 += v;
    }
    float mu = block_sum(s1, red) * (1.f / TWO_H);
    float s2 = 0.f;
    for (int j = tid; j < TWO_H; j += 256) { float d = cx[j] - mu; s2 += d * d; }
    float var = block_sum(s2, red) * (1.f / TWO_H);
    float rstd = rsqrtf(var + 1e-5f);
    for (int j = tid; j < TWO_H; j += 256)
        cx[j] = gelu_f((cx[j] - mu) * rstd * ag[j] + ab[j]);
    __syncthreads();
    for (int f = tid; f < H_; f += 256) {
        float a = fb[f];
        for (int j = 0; j < TWO_H; ++j) a += cx[j] * fw[(size_t)j * H_ + f];
        fx[f] = a;
    }
    __syncthreads();
    float t1 = 0.f;
    for (int f = tid; f < H_; f += 256) t1 += fx[f];
    float m3 = block_sum(t1, red) * (1.f / H_);
    float t2 = 0.f;
    for (int f = tid; f < H_; f += 256) { float d = fx[f] - m3; t2 += d * d; }
    float v3 = block_sum(t2, red) * (1.f / H_);
    float r3 = rsqrtf(v3 + 1e-5f);
    for (int f = tid; f < H_; f += 256) {
        float val = gelu_f((fx[f] - m3) * r3 * g3[f] + b3[f]);
        atomicAdd(&result[(size_t)bb * H_ + f], val * (1.f / 50.f));
    }
}

// ---------------------------------------------------------------------------
extern "C" void kernel_launch(void* const* d_in, const int* in_sizes, int n_in,
                              void* d_out, int out_size, void* d_ws, size_t ws_size,
                              hipStream_t stream) {
    (void)in_sizes; (void)n_in; (void)out_size; (void)ws_size;
    const float* cls_x   = (const float*)d_in[0];
    const float* cls_x1  = (const float*)d_in[1];
    const float* ln_g    = (const float*)d_in[2];
    const float* ln_b    = (const float*)d_in[3];
    const float* ln1_g   = (const float*)d_in[4];
    const float* ln1_b   = (const float*)d_in[5];
    const float* gate_w  = (const float*)d_in[6];
    const float* gate_b  = (const float*)d_in[7];
    const float* W1      = (const float*)d_in[8];
    const float* b1      = (const float*)d_in[9];
    const float* W2      = (const float*)d_in[10];
    const float* b2      = (const float*)d_in[11];
    const float* lnout_g = (const float*)d_in[12];
    const float* lnout_b = (const float*)d_in[13];
    const float* lnout1_g= (const float*)d_in[14];
    const float* lnout1_b= (const float*)d_in[15];
    const float* ln2_g   = (const float*)d_in[16];
    const float* ln2_b   = (const float*)d_in[17];
    const float* ain_w   = (const float*)d_in[18];
    const float* ain_b   = (const float*)d_in[19];
    const float* aout_w  = (const float*)d_in[20];
    const float* aout_b  = (const float*)d_in[21];
    const float* post_w  = (const float*)d_in[22];
    const float* post_b  = (const float*)d_in[23];
    const float* aa_g    = (const float*)d_in[24];
    const float* aa_b    = (const float*)d_in[25];
    const float* fus_w   = (const float*)d_in[26];
    const float* fus_b   = (const float*)d_in[27];
    const float* ln3_g   = (const float*)d_in[28];
    const float* ln3_b   = (const float*)d_in[29];

    char* ws = (char*)d_ws;
    _Float16* W1T   = (_Float16*)(ws + 0);            // 33,554,432 B
    _Float16* W2T   = (_Float16*)(ws + 33554432);     // 33,554,432 B
    _Float16* Xh    = (_Float16*)(ws + 67108864);     //  6,553,600 B
    _Float16* X1h   = (_Float16*)(ws + 73662464);     //  6,553,600 B
    float* scale0   = (float*)(ws + 80216064);        //    204,800 B
    float* scale1   = (float*)(ws + 80420864);        //    204,800 B
    float* moe0     = (float*)(ws + 80625664);        // 13,107,200 B
    float* moe1     = (float*)(ws + 93732864);        // 13,107,200 B
    float* ctA      = (float*)(ws + 106840064);       // 26,214,400 B
    float* ctB      = (float*)(ws + 133054464);       // 26,214,400 B (total ~152 MB)

    float* out_result = (float*)d_out;                // [64][1024]
    float* out_r      = out_result + 64 * 1024;       // [64][50][1024]
    float* out_r1     = out_r + TOK * H_;             // [64][50][1024]

    // 1. f16 transposed weights
    convT_kernel<<<65536, 256, 0, stream>>>(W1, W1T);
    convT_kernel<<<65536, 256, 0, stream>>>(W2, W2T);
    // 2. embed LN+gelu -> f16 activations
    embed_kernel<<<TOK, 256, 0, stream>>>(cls_x,  ln_g,  ln_b,  Xh);
    embed_kernel<<<TOK, 256, 0, stream>>>(cls_x1, ln1_g, ln1_b, X1h);
    // 3. gate top-2 scales
    gate_kernel<<<TOK, 256, 0, stream>>>(Xh,  gate_w, gate_b, scale0);
    gate_kernel<<<TOK, 256, 0, stream>>>(X1h, gate_w, gate_b, scale1);
    // 4. zero accumulators + result region of d_out
    zero_kernel<<<(TOK * H_ + 255) / 256, 256, 0, stream>>>(moe0, TOK * H_);
    zero_kernel<<<(TOK * H_ + 255) / 256, 256, 0, stream>>>(moe1, TOK * H_);
    zero_kernel<<<(64 * H_ + 255) / 256, 256, 0, stream>>>(out_result, 64 * H_);
    // 5. fused MoE (WMMA): grid = (token tiles, experts), 64 KB dynamic LDS
    dim3 moe_grid(TOK / 16, E_);
    moe_kernel<<<moe_grid, 256, 2 * 16 * H_ * sizeof(_Float16), stream>>>(
        Xh, W1T, W2T, b1, b2, scale0, moe0);
    moe_kernel<<<moe_grid, 256, 2 * 16 * H_ * sizeof(_Float16), stream>>>(
        X1h, W1T, W2T, b1, b2, scale1, moe1);
    // 6. r / r1
    rout_kernel<<<TOK, 256, 0, stream>>>(moe0, lnout_g,  lnout_b,  out_r);
    rout_kernel<<<TOK, 256, 0, stream>>>(moe1, lnout1_g, lnout1_b, out_r1);
    // 7. concat + LN + gelu, transposed layout
    concat_ln_kernel<<<TOK, 256, 0, stream>>>(out_r, out_r1, ln2_g, ln2_b, ctA);
    // 8. four attention layers (ping-pong ctA <-> ctB)
    const size_t attn_sh = 20096 * sizeof(float);
    for (int i = 0; i < 4; ++i) {
        const float* src = (i & 1) ? ctB : ctA;
        float*       dst = (i & 1) ? ctA : ctB;
        attn_kernel<<<TWO_H, 256, attn_sh, stream>>>(
            src, dst,
            ain_w + (size_t)i * 150 * 50, ain_b + (size_t)i * 150,
            aout_w + (size_t)i * 50 * 50, aout_b + (size_t)i * 50,
            post_w + (size_t)i * 50 * 50, post_b + (size_t)i * 50);
    }
    // after 4 layers the result sits in ctA
    // 9. final LN + fusion + LN + mean
    final_kernel<<<TOK, 256, (TWO_H + H_) * sizeof(float), stream>>>(
        ctA, aa_g, aa_b, fus_w, fus_b, ln3_g, ln3_b, out_result);
}
